// Attention_11407433138857
// MI455X (gfx1250) — matrix operand order repaired and checked by reading, not compiled, not run on hardware
//
#include <hip/hip_runtime.h>

typedef __attribute__((ext_vector_type(16))) _Float16 v16h;
typedef __attribute__((ext_vector_type(8)))  _Float16 v8h;
typedef __attribute__((ext_vector_type(4)))  _Float16 v4h;
typedef __attribute__((ext_vector_type(8)))  float    v8f;

#define B_SZ 2
#define TQ 2048
#define HQ 32
#define HKV 8
#define DH 128
#define GQA (HQ / HKV)
#define SCALE_F 0.08838834764831845f

#define BLOCK_M 64
#define BLOCK_N 64
#define NWAVES 4
#define KSTRIDE 136   // halves: 128 + 8 pad (16B-aligned rows, conflict-free B reads)
#define VTSTRIDE 72   // halves: 64 + 8 pad
#define PSTRIDE 72    // halves: 64 + 8 pad

__device__ __forceinline__ v16h pack16(v8h lo, v8h hi) {
  v16h r;
#pragma unroll
  for (int i = 0; i < 8; ++i) { r[i] = lo[i]; r[i + 8] = hi[i]; }
  return r;
}

// xor-style permute within each group of 16 lanes (keeps the two C-layout row
// halves independent). v_permlane16 (VALU) — confirmed selected on gfx1250.
__device__ __forceinline__ float lane16_xor(float x, unsigned sel0, unsigned sel1,
                                            int xmask) {
#if __has_builtin(__builtin_amdgcn_permlane16)
  unsigned u = __float_as_uint(x);
  u = __builtin_amdgcn_permlane16(u, u, sel0, sel1, false, false);
  return __uint_as_float(u);
#else
  (void)sel0; (void)sel1;
  return __shfl_xor(x, xmask, 32);
#endif
}
#define XOR1(x) lane16_xor((x), 0x67452301u, 0xEFCDAB89u, 1)
#define XOR2(x) lane16_xor((x), 0x54761032u, 0xDCFE98BAu, 2)
#define XOR4(x) lane16_xor((x), 0x32107654u, 0xBA98FEDCu, 4)
#define XOR8(x) lane16_xor((x), 0xFEDCBA98u, 0x76543210u, 8)

__global__ __launch_bounds__(NWAVES * 32)
void fa_fwd_kernel(const float* __restrict__ Q, const float* __restrict__ K,
                   const float* __restrict__ V, float* __restrict__ O)
{
  __shared__ __align__(16) _Float16 Ks[BLOCK_N * KSTRIDE];      // 17408 B
  __shared__ __align__(16) _Float16 Vt[DH * VTSTRIDE];          // 18432 B
  __shared__ __align__(16) _Float16 Ps[NWAVES * 16 * PSTRIDE];  //  9216 B

  const int tid   = threadIdx.x;
  const int wave  = tid >> 5;
  const int lane  = tid & 31;
  const int l16   = lane & 15;
  const int lhalf = lane >> 4;

  const int qtiles = TQ / BLOCK_M;
  const int bid = blockIdx.x;
  const int qt  = bid % qtiles;
  const int h   = (bid / qtiles) % HQ;
  const int b   = bid / (qtiles * HQ);
  const int hk  = h / GQA;

  const int q0     = qt * BLOCK_M;
  const int qwrow0 = q0 + wave * 16;   // first query row handled by this wave

  // ---- preload Q tile into WMMA A-layout registers (f16, pre-scaled) ----
  v16h qa[4];
  {
    const float* qp = Q + ((((size_t)b * TQ) + (size_t)(qwrow0 + l16)) * HQ + h) * DH;
    const int off = lhalf * 8;
#pragma unroll
    for (int c = 0; c < 4; ++c) {
      const float4 f0 = *(const float4*)(qp + c * 32 + off);
      const float4 f1 = *(const float4*)(qp + c * 32 + off + 4);
      const float4 f2 = *(const float4*)(qp + c * 32 + 16 + off);
      const float4 f3 = *(const float4*)(qp + c * 32 + 16 + off + 4);
      v16h a;
      a[0]  = (_Float16)(f0.x * SCALE_F); a[1]  = (_Float16)(f0.y * SCALE_F);
      a[2]  = (_Float16)(f0.z * SCALE_F); a[3]  = (_Float16)(f0.w * SCALE_F);
      a[4]  = (_Float16)(f1.x * SCALE_F); a[5]  = (_Float16)(f1.y * SCALE_F);
      a[6]  = (_Float16)(f1.z * SCALE_F); a[7]  = (_Float16)(f1.w * SCALE_F);
      a[8]  = (_Float16)(f2.x * SCALE_F); a[9]  = (_Float16)(f2.y * SCALE_F);
      a[10] = (_Float16)(f2.z * SCALE_F); a[11] = (_Float16)(f2.w * SCALE_F);
      a[12] = (_Float16)(f3.x * SCALE_F); a[13] = (_Float16)(f3.y * SCALE_F);
      a[14] = (_Float16)(f3.z * SCALE_F); a[15] = (_Float16)(f3.w * SCALE_F);
      qa[c] = a;
    }
  }

  // constant ones B-operand (32x16 of 1.0): row sums via the matrix pipe
  v16h ones;
#pragma unroll
  for (int i = 0; i < 16; ++i) ones[i] = (_Float16)1.0f;

  // ---- online-softmax state ----
  float m_run[8];
#pragma unroll
  for (int r = 0; r < 8; ++r) m_run[r] = -1e30f;
  v8f o_acc[8];
#pragma unroll
  for (int dt = 0; dt < 8; ++dt) o_acc[dt] = {};
  v8f o_sum = {};   // running row-sum accumulator (l), rescaled like o_acc

  // staging decompositions
  const int krow = tid >> 1;             // K: 0..63
  const int kc0  = (tid & 1) * 64;       // K: 0 or 64
  const int vng  = tid & 15;             // V: n-group -> rows 4*vng..4*vng+3
  const int vdg  = tid >> 4;             // V: d-group -> d 16*vdg..16*vdg+15

  for (int kb = 0; kb <= q0; kb += BLOCK_N) {
    // ---- cooperative K staging (row-major f16, vectorized 16B stores) ----
    {
      const size_t kvrow = (((size_t)b * TQ) + (size_t)(kb + krow)) * HKV + hk;
      const float* kp = K + kvrow * DH + kc0;
      if (kb < q0) {  // prefetch next tile's rows
        __builtin_prefetch(kp + (size_t)BLOCK_N * HKV * DH, 0, 1);
      }
#pragma unroll
      for (int j = 0; j < 64; j += 8) {
        const float4 a0 = *(const float4*)(kp + j);
        const float4 a1 = *(const float4*)(kp + j + 4);
        v8h hh;
        hh[0] = (_Float16)a0.x; hh[1] = (_Float16)a0.y;
        hh[2] = (_Float16)a0.z; hh[3] = (_Float16)a0.w;
        hh[4] = (_Float16)a1.x; hh[5] = (_Float16)a1.y;
        hh[6] = (_Float16)a1.z; hh[7] = (_Float16)a1.w;
        *(v8h*)&Ks[krow * KSTRIDE + kc0 + j] = hh;
      }
    }
    // ---- V^T staging: 4(n) x 16(d) sub-block per thread, packed b64 stores ----
    {
      const float* vbase =
          V + ((((size_t)b * TQ) + (size_t)(kb + 4 * vng)) * HKV + hk) * DH + 16 * vdg;
      if (kb < q0) {
        __builtin_prefetch(vbase + (size_t)BLOCK_N * HKV * DH, 0, 1);
      }
      _Float16 hv[4][16];
#pragma unroll
      for (int jn = 0; jn < 4; ++jn) {
        const float* vr = vbase + (size_t)jn * (HKV * DH);
#pragma unroll
        for (int jf = 0; jf < 4; ++jf) {
          const float4 x = *(const float4*)(vr + jf * 4);
          hv[jn][jf * 4 + 0] = (_Float16)x.x;
          hv[jn][jf * 4 + 1] = (_Float16)x.y;
          hv[jn][jf * 4 + 2] = (_Float16)x.z;
          hv[jn][jf * 4 + 3] = (_Float16)x.w;
        }
      }
#pragma unroll
      for (int jd = 0; jd < 16; ++jd) {
        v4h pk;
        pk[0] = hv[0][jd]; pk[1] = hv[1][jd];
        pk[2] = hv[2][jd]; pk[3] = hv[3][jd];
        *(v4h*)&Vt[(16 * vdg + jd) * VTSTRIDE + 4 * vng] = pk;
      }
    }
    __syncthreads();

    // ---- GEMM1: S = Q K^T  (four 16-key tiles, K-dim = 128 in 4 chunks) ----
    v8f s[4];
#pragma unroll
    for (int t = 0; t < 4; ++t) s[t] = {};
#pragma unroll
    for (int c = 0; c < 4; ++c) {
#pragma unroll
      for (int t = 0; t < 4; ++t) {
        const _Float16* bp = &Ks[(t * 16 + l16) * KSTRIDE + c * 32 + lhalf * 8];
        const v16h bb = pack16(*(const v8h*)bp, *(const v8h*)(bp + 16));
        s[t] = __builtin_amdgcn_wmma_f32_16x16x32_f16(false, qa[c], false, bb,
                                                      (short)0, s[t], false, false);
      }
    }

    // ---- causal mask (only the diagonal block needs it) ----
    if (kb + BLOCK_N - 1 > qwrow0) {
#pragma unroll
      for (int r = 0; r < 8; ++r) {
        const int mg = qwrow0 + r + 8 * lhalf;
#pragma unroll
        for (int t = 0; t < 4; ++t) {
          const int n = kb + t * 16 + l16;
          if (n > mg) s[t][r] = -1e30f;
        }
      }
    }

    // ---- online softmax: row max via half-wave xor reductions (permlane) ----
    float mnew[8], alpha[8];
#pragma unroll
    for (int r = 0; r < 8; ++r) {
      float bm = fmaxf(fmaxf(s[0][r], s[1][r]), fmaxf(s[2][r], s[3][r]));
      bm = fmaxf(bm, XOR1(bm));
      bm = fmaxf(bm, XOR2(bm));
      bm = fmaxf(bm, XOR4(bm));
      bm = fmaxf(bm, XOR8(bm));
      const float mn = fmaxf(m_run[r], bm);
      mnew[r]  = mn;
      alpha[r] = __expf(m_run[r] - mn);
      m_run[r] = mn;
    }
#pragma unroll
    for (int t = 0; t < 4; ++t)
#pragma unroll
      for (int r = 0; r < 8; ++r)
        s[t][r] = __expf(s[t][r] - mnew[r]);

    // rescale accumulators (incl. the running row-sum)
#pragma unroll
    for (int dt = 0; dt < 8; ++dt)
#pragma unroll
      for (int r = 0; r < 8; ++r)
        o_acc[dt][r] *= alpha[r];
#pragma unroll
    for (int r = 0; r < 8; ++r) o_sum[r] *= alpha[r];

    // ---- P: C-layout -> LDS -> A-layout (per-wave buffer, no barrier) ----
    _Float16* pw = &Ps[wave * 16 * PSTRIDE];
#pragma unroll
    for (int r = 0; r < 8; ++r) {
      const int m = r + 8 * lhalf;
#pragma unroll
      for (int t = 0; t < 4; ++t)
        pw[m * PSTRIDE + t * 16 + l16] = (_Float16)s[t][r];
    }
    const _Float16* pr = &Ps[wave * 16 * PSTRIDE + l16 * PSTRIDE + lhalf * 16];
    const v16h pa0 = pack16(*(const v8h*)pr,        *(const v8h*)(pr + 8));
    const v16h pa1 = pack16(*(const v8h*)(pr + 32), *(const v8h*)(pr + 40));

    // row sums via matrix pipe: o_sum += P * ones
    o_sum = __builtin_amdgcn_wmma_f32_16x16x32_f16(false, pa0, false, ones,
                                                   (short)0, o_sum, false, false);
    o_sum = __builtin_amdgcn_wmma_f32_16x16x32_f16(false, pa1, false, ones,
                                                   (short)0, o_sum, false, false);

    // ---- GEMM2: O += P V  (8 d-tiles of 16, K-dim = 64 keys in 2 chunks) ----
#pragma unroll
    for (int dt = 0; dt < 8; ++dt) {
      const _Float16* vb = &Vt[(dt * 16 + l16) * VTSTRIDE + lhalf * 16];
      const v16h b0 = pack16(*(const v8h*)vb,        *(const v8h*)(vb + 8));
      o_acc[dt] = __builtin_amdgcn_wmma_f32_16x16x32_f16(false, pa0, false, b0,
                                                         (short)0, o_acc[dt],
                                                         false, false);
      const v16h b1 = pack16(*(const v8h*)(vb + 32), *(const v8h*)(vb + 40));
      o_acc[dt] = __builtin_amdgcn_wmma_f32_16x16x32_f16(false, pa1, false, b1,
                                                         (short)0, o_acc[dt],
                                                         false, false);
    }
    __syncthreads();
  }

  // ---- epilogue: O / l, write fp32 output [B, Tq, Hq*D] ----
  {
    float* op = O + (((size_t)b * TQ + qwrow0) * (size_t)(HQ * DH)) + (size_t)h * DH;
#pragma unroll
    for (int r = 0; r < 8; ++r) {
      const int m = r + 8 * lhalf;
      const float inv = 1.0f / o_sum[r];
#pragma unroll
      for (int dt = 0; dt < 8; ++dt)
        op[(size_t)m * (HQ * DH) + dt * 16 + l16] = o_acc[dt][r] * inv;
    }
  }
}

extern "C" void kernel_launch(void* const* d_in, const int* in_sizes, int n_in,
                              void* d_out, int out_size, void* d_ws, size_t ws_size,
                              hipStream_t stream) {
  (void)in_sizes; (void)n_in; (void)out_size; (void)d_ws; (void)ws_size;
  const float* q = (const float*)d_in[0];
  const float* k = (const float*)d_in[1];
  const float* v = (const float*)d_in[2];
  float* out = (float*)d_out;
  const int grid = B_SZ * HQ * (TQ / BLOCK_M);  // 2048 workgroups
  fa_fwd_kernel<<<grid, NWAVES * 32, 0, stream>>>(q, k, v, out);
}
